// H2GCN_25555055411703
// MI455X (gfx1250) — matrix operand order repaired and checked by reading, not compiled, not run on hardware
//
#include <hip/hip_runtime.h>

// ---------------------------------------------------------------------------
// H2GCN forward for MI455X (gfx1250, wave32, WMMA).
//  - GEMMs via v_wmma_f32_16x16x32_f16; fp32 operands split into hi+lo f16
//    (3 WMMAs ~= fp32 product, f32 accumulate). Weights are pre-packed ONCE
//    into hi/lo f16 in the exact K-major layout the waves read, zero-padded
//    to 16-column tiles -> branch-free, conversion-free B path in the loop.
//  - A-side rows are clamped (not masked): row m only feeds C row m, and
//    stores are guarded, so no divergence around WMMA (EXEC stays all-ones).
//  - SpMM: rowptr via binary search on sorted COO rows, one wave per row,
//    lane = feature column; edge streams non-temporal so the 90MB feature
//    matrix stays resident in the 192MB L2 (HBM sees ~0.5GB total @23.3TB/s).
// ---------------------------------------------------------------------------

typedef __attribute__((ext_vector_type(16))) _Float16 v16h;
typedef __attribute__((ext_vector_type(8)))  float    v8f;

// Pack A-side: 8 fp32 at p0 (elements 0..7) + 8 fp32 at p1 (elements 8..15)
// into hi/lo f16 vectors.
__device__ __forceinline__ void packA(const float* p0, const float* p1,
                                      v16h& hi, v16h& lo)
{
    float f[16];
    const float4 q0 = ((const float4*)p0)[0];
    const float4 q1 = ((const float4*)p0)[1];
    const float4 q2 = ((const float4*)p1)[0];
    const float4 q3 = ((const float4*)p1)[1];
    f[0]=q0.x;  f[1]=q0.y;  f[2]=q0.z;  f[3]=q0.w;
    f[4]=q1.x;  f[5]=q1.y;  f[6]=q1.z;  f[7]=q1.w;
    f[8]=q2.x;  f[9]=q2.y;  f[10]=q2.z; f[11]=q2.w;
    f[12]=q3.x; f[13]=q3.y; f[14]=q3.z; f[15]=q3.w;
#pragma unroll
    for (int e = 0; e < 16; ++e) {
        _Float16 h = (_Float16)f[e];
        hi[e] = h;
        lo[e] = (_Float16)(f[e] - (float)h);
    }
}

// ---------------------------------------------------------------------------
// One-time weight pre-pack: W[ncols x K] fp32 -> hi/lo f16 [ncols_pad x K],
// rows >= ncols zero-filled (so the GEMM needs no B-side guard).
// ---------------------------------------------------------------------------
__global__ void h2gcn_pack_w(const float* __restrict__ W, int ncols, int K,
                             _Float16* __restrict__ hi, _Float16* __restrict__ lo,
                             int ncols_pad)
{
    const int i = blockIdx.x * blockDim.x + threadIdx.x;
    if (i >= ncols_pad * K) return;
    const int n = i / K;
    const float f = (n < ncols) ? W[i] : 0.f;   // W shares the n*K+k indexing
    const _Float16 h = (_Float16)f;
    hi[i] = h;
    lo[i] = (_Float16)(f - (float)h);
}

// ---------------------------------------------------------------------------
// rowptr[i] = lower_bound(row, i); rows are sorted (np.unique in reference).
// ---------------------------------------------------------------------------
__global__ void h2gcn_build_rowptr(const int* __restrict__ row, int E,
                                   int* __restrict__ rowptr, int n)
{
    int i = blockIdx.x * blockDim.x + threadIdx.x;
    if (i > n) return;
    int lo = 0, hi = E;
    while (lo < hi) {
        int mid = (lo + hi) >> 1;
        if (row[mid] < i) lo = mid + 1; else hi = mid;
    }
    rowptr[i] = lo;
}

// ---------------------------------------------------------------------------
// WMMA GEMM:  Y[m, n] = act( sum_k A[m,k] * W[n,k] + bias[n] )
//   A: M x K fp32 row-major (lda), K % 32 == 0.
//   Whi/Wlo: pre-packed f16 [NT*16 x K] K-major (zero-padded columns).
//   One wave per 16-row tile, NT 16-col tiles accumulated per wave.
//   WMMAs are interleaved across the NT independent accumulators.
// ---------------------------------------------------------------------------
template<int NT, bool RELU>
__global__ __launch_bounds__(256)
void h2gcn_wmma_gemm(const float* __restrict__ A, int M, int K, int lda,
                     const _Float16* __restrict__ Whi,
                     const _Float16* __restrict__ Wlo,
                     int ncols, const float* __restrict__ bias,
                     float* __restrict__ Y, int ldy)
{
    const int lane = threadIdx.x & 31;
    const int wid  = threadIdx.x >> 5;
    const int m0   = (blockIdx.x * 8 + wid) * 16;
    if (m0 >= M) return;                 // uniform whole-wave exit

    const int h  = lane >> 4;            // lane half (selects K sub-block)
    const int nl = lane & 15;

    v8f c[NT];
#pragma unroll
    for (int t = 0; t < NT; ++t) c[t] = (v8f){0,0,0,0,0,0,0,0};

    // A layout (16-bit, 16x32): lane%16 = M row; lane half h:
    //   elements 0..15 -> K = kb + 16h + e (matches B)
    // Clamp (not mask) the row: garbage only reaches C rows never stored.
    int mrow = m0 + nl;
    if (mrow >= M) mrow = M - 1;
    const float* arow = A + (size_t)mrow * lda;

    // B layout (16-bit, 32x16): lane%16 = N column; elements -> K = kb+16h+e,
    // i.e. 16 contiguous pre-packed halves (32B aligned).
    const size_t boff = (size_t)nl * K + 16 * h;

    for (int kb = 0; kb < K; kb += 32) {
        v16h ahi, alo;
        packA(arow + kb + 16 * h, arow + kb + 16 * h + 8, ahi, alo);

        v16h bhi[NT], blo[NT];
#pragma unroll
        for (int t = 0; t < NT; ++t) {
            bhi[t] = *(const v16h*)(Whi + boff + (size_t)t * 16 * K + kb);
            blo[t] = *(const v16h*)(Wlo + boff + (size_t)t * 16 * K + kb);
        }
#pragma unroll
        for (int t = 0; t < NT; ++t)
            c[t] = __builtin_amdgcn_wmma_f32_16x16x32_f16(
                       false, ahi, false, bhi[t], (short)0, c[t], false, false);
#pragma unroll
        for (int t = 0; t < NT; ++t)
            c[t] = __builtin_amdgcn_wmma_f32_16x16x32_f16(
                       false, alo, false, bhi[t], (short)0, c[t], false, false);
#pragma unroll
        for (int t = 0; t < NT; ++t)
            c[t] = __builtin_amdgcn_wmma_f32_16x16x32_f16(
                       false, ahi, false, blo[t], (short)0, c[t], false, false);
    }

    // C/D layout: lane%16 = N column; VGPR r -> M row = m0 + 8h + r
#pragma unroll
    for (int t = 0; t < NT; ++t) {
        const int n = t * 16 + nl;
        if (n >= ncols) continue;
        const float bb = bias[n];
#pragma unroll
        for (int r = 0; r < 8; ++r) {
            const int m = m0 + 8 * h + r;
            if (m < M) {
                float v = c[t][r] + bb;
                if (RELU) v = v > 0.f ? v : 0.f;
                Y[(size_t)m * ldy + n] = v;
            }
        }
    }
}

// ---------------------------------------------------------------------------
// SpMM: one wave per output row, lane = feature lane (F = 64 or 128).
//   xout[i, f] = sum_e val[e] * xin[col[e], f]   over e in CSR row i.
// Edge streams non-temporal (used once/layer); feature gathers hit L2.
// 2-edge unroll for gather-latency ILP.
// ---------------------------------------------------------------------------
__global__ __launch_bounds__(256)
void h2gcn_spmm_row(const int* __restrict__ rowptr, const int* __restrict__ col,
                    const float* __restrict__ val,
                    const float* __restrict__ xin, float* __restrict__ xout,
                    int nrows, int F, int ld)
{
    const int wave = blockIdx.x * (blockDim.x >> 5) + (threadIdx.x >> 5);
    const int lane = threadIdx.x & 31;
    if (wave >= nrows) return;

    const int beg = rowptr[wave];
    const int end = rowptr[wave + 1];
    const int nf  = F >> 5;              // 2 (F=64) or 4 (F=128)

    float acc0 = 0.f, acc1 = 0.f, acc2 = 0.f, acc3 = 0.f;
    int e = beg;
    for (; e + 1 < end; e += 2) {
        const int   c0 = __builtin_nontemporal_load(col + e);
        const int   c1 = __builtin_nontemporal_load(col + e + 1);
        const float v0 = __builtin_nontemporal_load(val + e);
        const float v1 = __builtin_nontemporal_load(val + e + 1);
        const float* x0 = xin + (size_t)c0 * ld + lane;
        const float* x1 = xin + (size_t)c1 * ld + lane;
        acc0 += v0 * x0[0];
        if (nf > 1) acc1 += v0 * x0[32];
        if (nf > 2) acc2 += v0 * x0[64];
        if (nf > 3) acc3 += v0 * x0[96];
        acc0 += v1 * x1[0];
        if (nf > 1) acc1 += v1 * x1[32];
        if (nf > 2) acc2 += v1 * x1[64];
        if (nf > 3) acc3 += v1 * x1[96];
    }
    if (e < end) {
        const int   c0 = __builtin_nontemporal_load(col + e);
        const float v0 = __builtin_nontemporal_load(val + e);
        const float* x0 = xin + (size_t)c0 * ld + lane;
        acc0 += v0 * x0[0];
        if (nf > 1) acc1 += v0 * x0[32];
        if (nf > 2) acc2 += v0 * x0[64];
        if (nf > 3) acc3 += v0 * x0[96];
    }
    float* yr = xout + (size_t)wave * ld + lane;
    yr[0] = acc0;
    if (nf > 1) yr[32] = acc1;
    if (nf > 2) yr[64] = acc2;
    if (nf > 3) yr[96] = acc3;
}

// ---------------------------------------------------------------------------
// Workspace: X[N x 448] concat buffer, packed weights, rowptrs (~90MB total).
// Column map: [0:64]=h0, [64:128]=L1 1-hop, [128:192]=L1 2-hop,
// [192:320]=L2 1-hop, [320:448]=L2 2-hop — the reference concat order.
// ---------------------------------------------------------------------------
extern "C" void kernel_launch(void* const* d_in, const int* in_sizes, int n_in,
                              void* d_out, int out_size, void* d_ws, size_t ws_size,
                              hipStream_t stream)
{
    const float* x    = (const float*)d_in[0];
    const float* W1   = (const float*)d_in[1];
    const float* b1   = (const float*)d_in[2];
    const float* Wf   = (const float*)d_in[3];
    const float* bf   = (const float*)d_in[4];
    const int*   row1 = (const int*)  d_in[5];
    const int*   col1 = (const int*)  d_in[6];
    const float* val1 = (const float*)d_in[7];
    const int*   row2 = (const int*)  d_in[8];
    const int*   col2 = (const int*)  d_in[9];
    const float* val2 = (const float*)d_in[10];

    const int E1 = in_sizes[5];
    const int E2 = in_sizes[8];
    const int N  = in_sizes[0] / 128;    // 50000
    const int LD = 448;                  // concat width

    // --- workspace carve-up (all f16 blocks 32B aligned) ---
    char* ws = (char*)d_ws;
    size_t off = 0;
    float* X = (float*)(ws + off);              off += (size_t)N * LD * sizeof(float);
    _Float16* W1hi = (_Float16*)(ws + off);     off += (size_t)64 * 128 * sizeof(_Float16);
    _Float16* W1lo = (_Float16*)(ws + off);     off += (size_t)64 * 128 * sizeof(_Float16);
    _Float16* Wfhi = (_Float16*)(ws + off);     off += (size_t)48 * 448 * sizeof(_Float16);
    _Float16* Wflo = (_Float16*)(ws + off);     off += (size_t)48 * 448 * sizeof(_Float16);
    int* rowptr1 = (int*)(ws + off);            off += (size_t)(N + 1) * sizeof(int);
    int* rowptr2 = (int*)(ws + off);

    // 1. One-time weight packing + CSR row pointers.
    h2gcn_pack_w<<<(64 * 128 + 255) / 256, 256, 0, stream>>>(W1, 64, 128, W1hi, W1lo, 64);
    h2gcn_pack_w<<<(48 * 448 + 255) / 256, 256, 0, stream>>>(Wf, 40, 448, Wfhi, Wflo, 48);
    {
        const int tb = 256, gb = (N + 1 + tb - 1) / tb;
        h2gcn_build_rowptr<<<gb, tb, 0, stream>>>(row1, E1, rowptr1, N);
        h2gcn_build_rowptr<<<gb, tb, 0, stream>>>(row2, E2, rowptr2, N);
    }

    const int mtiles      = (N + 15) / 16;        // 3125
    const int gemm_blocks = (mtiles + 7) / 8;     // 8 waves / block

    // 2. h0 = relu(x @ W1^T + b1) -> X[:, 0:64]
    h2gcn_wmma_gemm<4, true><<<gemm_blocks, 256, 0, stream>>>(
        x, N, 128, 128, W1hi, W1lo, 64, b1, X, LD);

    // 3. Layer 1: aggregate h0 (F=64) over 1-hop and strict-2-hop graphs.
    const int sp_blocks = (N + 7) / 8;            // 8 waves / block
    h2gcn_spmm_row<<<sp_blocks, 256, 0, stream>>>(rowptr1, col1, val1, X,      X + 64,  N, 64,  LD);
    h2gcn_spmm_row<<<sp_blocks, 256, 0, stream>>>(rowptr2, col2, val2, X,      X + 128, N, 64,  LD);

    // 4. Layer 2: aggregate layer-1 output (F=128).
    h2gcn_spmm_row<<<sp_blocks, 256, 0, stream>>>(rowptr1, col1, val1, X + 64, X + 192, N, 128, LD);
    h2gcn_spmm_row<<<sp_blocks, 256, 0, stream>>>(rowptr2, col2, val2, X + 64, X + 320, N, 128, LD);

    // 5. out = X @ Wf^T + bf   (40 cols -> 3 tiles, B pre-padded to 48)
    h2gcn_wmma_gemm<3, false><<<gemm_blocks, 256, 0, stream>>>(
        X, N, 448, 448, Wfhi, Wflo, 40, bf, (float*)d_out, 40);
}
